// GraphSAGEClassifier_78649441124639
// MI455X (gfx1250) — matrix operand order repaired and checked by reading, not compiled, not run on hardware
//
#include <hip/hip_runtime.h>
#include <stdint.h>

// ---------------------------------------------------------------------------
// Problem dims
// ---------------------------------------------------------------------------
// B=64, W=48, BW=3072, N=400 (=25*16), NP=416 (K-padded to mult of 32), H=64, C=2
// Chunked over batches: 4 chunks x 16 batches (768 windows each) to bound ws.

typedef __attribute__((ext_vector_type(16))) _Float16 v16h;
typedef __attribute__((ext_vector_type(8)))  float    v8f;

union FragU { v16h v; uint32_t u[8]; };
union HPack { uint32_t u; _Float16 h[2]; };

// A-fragment (16x32 f16): lane holds row M=l&15; caller passes pointer to
// (row, kbase + (lane>>4)*16); 16 contiguous halves.
__device__ inline v16h load_fragA(const _Float16* p) {
  FragU f;
  const uint32_t* q = (const uint32_t*)p;
  f.u[0] = q[0]; f.u[1] = q[1]; f.u[2] = q[2]; f.u[3] = q[3];
  f.u[4] = q[4]; f.u[5] = q[5]; f.u[6] = q[6]; f.u[7] = q[7];
  return f.v;
}

// B-fragment (32x16 f16): lane holds column N=l&15; caller passes pointer to
// (colrow, kbase + (lane>>4)*16); 16 contiguous halves.
__device__ inline v16h load_fragB(const _Float16* p) {
  FragU f;
  const uint32_t* q = (const uint32_t*)p;
#pragma unroll
  for (int i = 0; i < 8; ++i) f.u[i] = q[i];
  return f.v;
}

__device__ inline v8f wmma16(v16h a, v16h b, v8f c) {
  return __builtin_amdgcn_wmma_f32_16x16x32_f16(false, a, false, b, (short)0, c,
                                                false, false);
}

// ---------------------------------------------------------------------------
// Workspace layout
// ---------------------------------------------------------------------------
constexpr size_t A256(size_t x) { return (x + 255) & ~(size_t)255; }
constexpr size_t O_DINV  = 0;                                        // 400 f32
constexpr size_t O_P16   = A256(O_DINV  + 400 * 4);                  // 400x416 f16
constexpr size_t O_X16   = A256(O_P16   + (size_t)400 * 416 * 2);    // 3072x416 f16
constexpr size_t O_WA2H  = A256(O_X16   + (size_t)3072 * 416 * 2);   // 64x64 f16
constexpr size_t O_WS2H  = A256(O_WA2H  + 64 * 64 * 2);              // 64x64 f16
constexpr size_t O_B2    = A256(O_WS2H  + 64 * 64 * 2);              // 64 f32 (ba2+bs2)
constexpr size_t O_NAGG1 = A256(O_B2    + 64 * 4);                   // 768x400 f32
constexpr size_t O_H1    = A256(O_NAGG1 + (size_t)768 * 400 * 4);    // 768x400x64 f16
constexpr size_t O_NAGG2 = A256(O_H1    + (size_t)768 * 400 * 64 * 2);
constexpr size_t O_HMEAN = A256(O_NAGG2 + (size_t)768 * 400 * 64 * 2); // 64x400x64 f32

// ---------------------------------------------------------------------------
// Prep kernels
// ---------------------------------------------------------------------------
__global__ __launch_bounds__(64) void k_prep_dinv(const float* __restrict__ adj,
                                                  float* __restrict__ dinv) {
  __shared__ float red[64];
  int i = blockIdx.x, tid = threadIdx.x;
  float s = 0.f;
  for (int j = tid; j < 400; j += 64)
    s += adj[(size_t)i * 400 + j] + (j == i ? 1.f : 0.f);
  red[tid] = s; __syncthreads();
  for (int st = 32; st > 0; st >>= 1) {
    if (tid < st) red[tid] += red[tid + st];
    __syncthreads();
  }
  if (tid == 0) { float d = red[0]; dinv[i] = d > 0.f ? rsqrtf(d) : 0.f; }
}

__global__ __launch_bounds__(64) void k_prep_P(const float* __restrict__ adj,
                                               const float* __restrict__ dinv,
                                               _Float16* __restrict__ P16) {
  __shared__ float red[64];
  __shared__ float invdeg;
  int i = blockIdx.x, tid = threadIdx.x;
  float di = dinv[i];
  float s = 0.f;
  for (int j = tid; j < 400; j += 64)
    s += (adj[(size_t)i * 400 + j] + (j == i ? 1.f : 0.f)) * dinv[j];
  red[tid] = s; __syncthreads();
  for (int st = 32; st > 0; st >>= 1) {
    if (tid < st) red[tid] += red[tid + st];
    __syncthreads();
  }
  if (tid == 0) {
    float deg = fmaxf(di * red[0], 1.f);
    invdeg = di / deg;
  }
  __syncthreads();
  float sc = invdeg;
  for (int j = tid; j < 416; j += 64) {
    float v = 0.f;
    if (j < 400)
      v = (adj[(size_t)i * 400 + j] + (j == i ? 1.f : 0.f)) * dinv[j] * sc;
    P16[(size_t)i * 416 + j] = (_Float16)v;
  }
}

__global__ __launch_bounds__(256) void k_prep_x(const float* __restrict__ bold,
                                                _Float16* __restrict__ X16) {
  size_t idx = (size_t)blockIdx.x * 256 + threadIdx.x;  // < 3072*416
  int r = (int)(idx / 416), c = (int)(idx % 416);
  X16[idx] = (c < 400) ? (_Float16)bold[(size_t)r * 400 + c] : (_Float16)0.f;
}

__global__ __launch_bounds__(256) void k_prep_w(const float* __restrict__ Wa2,
                                                const float* __restrict__ Ws2,
                                                const float* __restrict__ ba2,
                                                const float* __restrict__ bs2,
                                                _Float16* __restrict__ Wa2h,
                                                _Float16* __restrict__ Ws2h,
                                                float* __restrict__ bias2) {
  int tid = threadIdx.x;
  for (int i = tid; i < 4096; i += 256) {
    Wa2h[i] = (_Float16)Wa2[i];
    Ws2h[i] = (_Float16)Ws2[i];
  }
  if (tid < 64) bias2[tid] = ba2[tid] + bs2[tid];
}

// ---------------------------------------------------------------------------
// Layer-1 aggregation GEMM: nagg1[768,400] = X16[768,416] * P16^T (B[k][n]=P[n][k])
// Block: 256 thr = 8 waves; tile M=128 (wave m-tile), N=80 (5 n-tiles/wave), K=416.
// ---------------------------------------------------------------------------
__global__ __launch_bounds__(256) void k_l1gemm(const _Float16* __restrict__ X16,
                                                const _Float16* __restrict__ P16,
                                                float* __restrict__ nagg1c,
                                                int chunk) {
  int mb = (blockIdx.x / 5) * 128;   // local row base (0..640)
  int nb = (blockIdx.x % 5) * 80;
  int wave = threadIdx.x >> 5, lane = threadIdx.x & 31;
  int kgA = (lane >> 4) * 16, kgB = (lane >> 4) * 16;
  int mrow = mb + wave * 16 + (lane & 15);
  const _Float16* aptr = X16 + (size_t)(chunk * 768 + mrow) * 416 + kgA;

  v8f acc[5] = {};
  for (int ks = 0; ks < 13; ++ks) {
    v16h a = load_fragA(aptr + ks * 32);
#pragma unroll
    for (int t = 0; t < 5; ++t) {
      int n = nb + t * 16 + (lane & 15);
      v16h b = load_fragB(P16 + (size_t)n * 416 + ks * 32 + kgB);
      acc[t] = wmma16(a, b, acc[t]);
    }
  }
  int hi = (lane >> 4) * 8;
#pragma unroll
  for (int t = 0; t < 5; ++t)
#pragma unroll
    for (int r = 0; r < 8; ++r) {
      int row = mb + wave * 16 + r + hi;
      int col = nb + t * 16 + (lane & 15);
      nagg1c[(size_t)row * 400 + col] = acc[t][r];
    }
}

// ---------------------------------------------------------------------------
// Layer-1 pointwise: h1 = relu(LN(nagg1*Wa1 + x*Ws1 + ba1+bs1)), f16 out.
// One thread per (bw,n) row of 64 channels.
// ---------------------------------------------------------------------------
__global__ __launch_bounds__(256) void k_l1pw(const float* __restrict__ bold,
                                              const float* __restrict__ nagg1c,
                                              const float* __restrict__ Wa1,
                                              const float* __restrict__ ba1,
                                              const float* __restrict__ Ws1,
                                              const float* __restrict__ bs1,
                                              const float* __restrict__ g1,
                                              const float* __restrict__ b1,
                                              _Float16* __restrict__ h1c,
                                              int chunk) {
  __shared__ float wA[64], wS[64], bsum[64], gg[64], bb[64];
  if (threadIdx.x < 64) {
    int h = threadIdx.x;
    wA[h] = Wa1[h]; wS[h] = Ws1[h];
    bsum[h] = ba1[h] + bs1[h];
    gg[h] = g1[h]; bb[h] = b1[h];
  }
  __syncthreads();
  size_t r = (size_t)blockIdx.x * 256 + threadIdx.x;  // < 307200
  float na = nagg1c[r];
  float xv = bold[(size_t)chunk * 307200 + r];
  float s = 0.f, s2 = 0.f;
#pragma unroll 8
  for (int h = 0; h < 64; ++h) {
    float v = na * wA[h] + xv * wS[h] + bsum[h];
    s += v; s2 += v * v;
  }
  float m = s * (1.f / 64.f);
  float var = s2 * (1.f / 64.f) - m * m;
  float rstd = rsqrtf(var + 1e-5f);
  uint32_t* out = (uint32_t*)(h1c + r * 64);
#pragma unroll 8
  for (int h = 0; h < 64; h += 2) {
    float v0 = na * wA[h] + xv * wS[h] + bsum[h];
    float v1 = na * wA[h + 1] + xv * wS[h + 1] + bsum[h + 1];
    float o0 = fmaxf((v0 - m) * rstd * gg[h] + bb[h], 0.f);
    float o1 = fmaxf((v1 - m) * rstd * gg[h + 1] + bb[h + 1], 0.f);
    HPack pk; pk.h[0] = (_Float16)o0; pk.h[1] = (_Float16)o1;
    out[h >> 1] = pk.u;
  }
}

// ---------------------------------------------------------------------------
// Layer-2 aggregation GEMM (dominant op): per window, nagg2 = P[400,400] * h1[400,64]
// Block: 128 thr = 4 waves; tile M=80 nodes x N=64 ch, K=416.
// h1 transposed into LDS (K-major for B frags); P A-frags streamed from L2.
// ---------------------------------------------------------------------------
__global__ __launch_bounds__(128) void k_l2gemm(const _Float16* __restrict__ h1c,
                                                const _Float16* __restrict__ P16,
                                                _Float16* __restrict__ nagg2c) {
  __shared__ __align__(16) _Float16 h1T[64 * 420];  // [ch][node], padded pitch
  int bwl = blockIdx.x / 5;
  int mb  = (blockIdx.x % 5) * 80;
  int tid = threadIdx.x;

  // Stage h1[bwl] (row-major [400][64]) transposed into LDS.
  const uint32_t* s32 = (const uint32_t*)(h1c + (size_t)bwl * 25600);
  for (int idx = tid; idx < 12800; idx += 128) {
    int j = idx >> 5, cd = idx & 31;
    HPack pk; pk.u = s32[idx];
    h1T[(2 * cd + 0) * 420 + j] = pk.h[0];
    h1T[(2 * cd + 1) * 420 + j] = pk.h[1];
  }
  for (int idx = tid; idx < 64 * 16; idx += 128) {   // zero K pad 400..415
    int ch = idx >> 4, j = 400 + (idx & 15);
    h1T[ch * 420 + j] = (_Float16)0.f;
  }
  __syncthreads();

  int wave = tid >> 5, lane = tid & 31;
  int kgA = (lane >> 4) * 16, kgB = (lane >> 4) * 16;
  int cb = wave * 16;
  const _Float16* brow = h1T + (cb + (lane & 15)) * 420 + kgB;
  const _Float16* arow = P16 + (size_t)(mb + (lane & 15)) * 416 + kgA;

  v8f acc[5] = {};
  for (int ks = 0; ks < 13; ++ks) {
    v16h b = load_fragB(brow + ks * 32);
#pragma unroll
    for (int m = 0; m < 5; ++m) {
      v16h a = load_fragA(arow + (size_t)m * 16 * 416 + ks * 32);
      acc[m] = wmma16(a, b, acc[m]);
    }
  }

  int hi = (lane >> 4) * 8;
  _Float16* dst = nagg2c + (size_t)bwl * 25600;
#pragma unroll
  for (int m = 0; m < 5; ++m)
#pragma unroll
    for (int r = 0; r < 8; ++r) {
      int node = mb + m * 16 + r + hi;
      int ch = cb + (lane & 15);
      dst[(size_t)node * 64 + ch] = (_Float16)acc[m][r];
    }
}

// ---------------------------------------------------------------------------
// Layer-2 epilogue: h2 = relu(LN(nagg2*Wa2^T + h1*Ws2^T + bias)); mean over W.
// Block = 1 wave, handles (batch, 16-node tile), loops 48 windows.
// ---------------------------------------------------------------------------
__global__ __launch_bounds__(32) void k_l2epi(const _Float16* __restrict__ nagg2c,
                                              const _Float16* __restrict__ h1c,
                                              const _Float16* __restrict__ Wa2h,
                                              const _Float16* __restrict__ Ws2h,
                                              const float* __restrict__ bias2,
                                              const float* __restrict__ g2,
                                              const float* __restrict__ b2,
                                              float* __restrict__ hmean,
                                              int chunk) {
  __shared__ __align__(16) _Float16 Wt[64 * 68], St[64 * 68];
  __shared__ __align__(16) _Float16 An[16 * 68], Ah[16 * 68];
  __shared__ float Dt[16 * 66];
  __shared__ float Mac[16 * 64];
  __shared__ float gsh[64], bsh[64];

  int bl = blockIdx.x / 25;        // batch within chunk
  int n0 = (blockIdx.x % 25) * 16; // node tile base
  int tid = threadIdx.x;           // 0..31 (== lane)

  const uint32_t* wsrc = (const uint32_t*)Wa2h;
  const uint32_t* ssrc = (const uint32_t*)Ws2h;
  uint32_t* wdst = (uint32_t*)Wt;
  uint32_t* sdst = (uint32_t*)St;
  for (int idx = tid; idx < 2048; idx += 32) {
    int r = idx >> 5, cdw = idx & 31;
    wdst[r * 34 + cdw] = wsrc[idx];
    sdst[r * 34 + cdw] = ssrc[idx];
  }
  for (int idx = tid; idx < 1024; idx += 32) Mac[idx] = 0.f;
  if (tid < 64) { gsh[tid] = g2[tid]; bsh[tid] = b2[tid]; }
  float bv[4];
#pragma unroll
  for (int t = 0; t < 4; ++t) bv[t] = bias2[t * 16 + (tid & 15)];
  __syncthreads();

  int kgA = (tid >> 4) * 16, kgB = (tid >> 4) * 16;
  int ml = tid & 15, hi = (tid >> 4) * 8;
  uint32_t* An32 = (uint32_t*)An;
  uint32_t* Ah32 = (uint32_t*)Ah;

  for (int w = 0; w < 48; ++w) {
    size_t base = ((size_t)(bl * 48 + w) * 400 + n0) * 64;  // halves (even)
    const uint32_t* sn = (const uint32_t*)nagg2c + base / 2;
    const uint32_t* sh = (const uint32_t*)h1c + base / 2;
    for (int idx = tid; idx < 512; idx += 32) {
      int r = idx >> 5, cdw = idx & 31;
      An32[r * 34 + cdw] = sn[idx];
      Ah32[r * 34 + cdw] = sh[idx];
    }
    __syncthreads();

    v16h an0 = load_fragA(An + ml * 68 + 0 + kgA);
    v16h an1 = load_fragA(An + ml * 68 + 32 + kgA);
    v16h ah0 = load_fragA(Ah + ml * 68 + 0 + kgA);
    v16h ah1 = load_fragA(Ah + ml * 68 + 32 + kgA);

#pragma unroll
    for (int t = 0; t < 4; ++t) {
      int cl = t * 16 + ml;
      v8f acc;
#pragma unroll
      for (int i = 0; i < 8; ++i) acc[i] = bv[t];
      acc = wmma16(an0, load_fragB(Wt + cl * 68 + 0 + kgB), acc);
      acc = wmma16(an1, load_fragB(Wt + cl * 68 + 32 + kgB), acc);
      acc = wmma16(ah0, load_fragB(St + cl * 68 + 0 + kgB), acc);
      acc = wmma16(ah1, load_fragB(St + cl * 68 + 32 + kgB), acc);
#pragma unroll
      for (int r = 0; r < 8; ++r) Dt[(r + hi) * 66 + cl] = acc[r];
    }
    __syncthreads();

    if (tid < 16) {
      float s = 0.f, s2 = 0.f;
      for (int h = 0; h < 64; ++h) {
        float v = Dt[tid * 66 + h];
        s += v; s2 += v * v;
      }
      float m = s * (1.f / 64.f);
      float var = s2 * (1.f / 64.f) - m * m;
      float rstd = rsqrtf(var + 1e-5f);
      for (int h = 0; h < 64; ++h) {
        float v = fmaxf((Dt[tid * 66 + h] - m) * rstd * gsh[h] + bsh[h], 0.f);
        Mac[tid * 64 + h] += v;
      }
    }
    __syncthreads();
  }

  int b = chunk * 16 + bl;
  for (int idx = tid; idx < 1024; idx += 32) {
    int r = idx >> 6, h = idx & 63;
    hmean[((size_t)b * 400 + n0 + r) * 64 + h] = Mac[idx] * (1.f / 48.f);
  }
}

// ---------------------------------------------------------------------------
// Attention readout + head. One block per batch.
// ---------------------------------------------------------------------------
__global__ __launch_bounds__(256) void k_head(const float* __restrict__ hmean,
                                              const float* __restrict__ Wat,
                                              const float* __restrict__ bat,
                                              const float* __restrict__ Wh1,
                                              const float* __restrict__ bh1,
                                              const float* __restrict__ gh,
                                              const float* __restrict__ bh,
                                              const float* __restrict__ Wh2,
                                              const float* __restrict__ bh2,
                                              float* __restrict__ out) {
  __shared__ float sc[400];
  __shared__ float red[256];
  __shared__ float pooled[64];
  __shared__ float zz[64];
  __shared__ float st2[2];
  int b = blockIdx.x, tid = threadIdx.x;
  const float* hb = hmean + (size_t)b * 25600;

  for (int n = tid; n < 400; n += 256) {
    float a = bat[0];
    for (int h = 0; h < 64; ++h) a += hb[(size_t)n * 64 + h] * Wat[h];
    sc[n] = a;
  }
  __syncthreads();

  float lm = -1e30f;
  for (int n = tid; n < 400; n += 256) lm = fmaxf(lm, sc[n]);
  red[tid] = lm; __syncthreads();
  for (int st = 128; st > 0; st >>= 1) {
    if (tid < st) red[tid] = fmaxf(red[tid], red[tid + st]);
    __syncthreads();
  }
  float M = red[0];
  __syncthreads();

  float ls = 0.f;
  for (int n = tid; n < 400; n += 256) {
    float e = __expf(sc[n] - M);
    sc[n] = e; ls += e;
  }
  red[tid] = ls; __syncthreads();
  for (int st = 128; st > 0; st >>= 1) {
    if (tid < st) red[tid] += red[tid + st];
    __syncthreads();
  }
  float S = red[0];
  __syncthreads();

  if (tid < 64) {
    float a = 0.f;
    for (int n = 0; n < 400; ++n) a += sc[n] * hb[(size_t)n * 64 + tid];
    pooled[tid] = a / S;
  }
  __syncthreads();
  if (tid < 64) {
    float a = bh1[tid];
    for (int k = 0; k < 64; ++k) a += pooled[k] * Wh1[tid * 64 + k];
    zz[tid] = a;
  }
  __syncthreads();
  if (tid == 0) {
    float s = 0.f, s2 = 0.f;
    for (int h = 0; h < 64; ++h) { s += zz[h]; s2 += zz[h] * zz[h]; }
    float m = s * (1.f / 64.f);
    st2[0] = m;
    st2[1] = rsqrtf(s2 * (1.f / 64.f) - m * m + 1e-5f);
  }
  __syncthreads();
  if (tid < 2) {
    float a = bh2[tid];
    for (int h = 0; h < 64; ++h) {
      float v = fmaxf((zz[h] - st2[0]) * st2[1] * gh[h] + bh[h], 0.f);
      a += v * Wh2[tid * 64 + h];
    }
    out[b * 2 + tid] = a;
  }
}

// ---------------------------------------------------------------------------
// Launch
// ---------------------------------------------------------------------------
extern "C" void kernel_launch(void* const* d_in, const int* in_sizes, int n_in,
                              void* d_out, int out_size, void* d_ws, size_t ws_size,
                              hipStream_t stream) {
  (void)in_sizes; (void)n_in; (void)out_size; (void)ws_size;
  const float* bold = (const float*)d_in[0];
  const float* adj  = (const float*)d_in[1];
  const float* Wa1 = (const float*)d_in[2];
  const float* ba1 = (const float*)d_in[3];
  const float* Ws1 = (const float*)d_in[4];
  const float* bs1 = (const float*)d_in[5];
  const float* g1  = (const float*)d_in[6];
  const float* b1  = (const float*)d_in[7];
  const float* Wa2 = (const float*)d_in[8];
  const float* ba2 = (const float*)d_in[9];
  const float* Ws2 = (const float*)d_in[10];
  const float* bs2 = (const float*)d_in[11];
  const float* g2  = (const float*)d_in[12];
  const float* b2  = (const float*)d_in[13];
  const float* Wat = (const float*)d_in[14];
  const float* bat = (const float*)d_in[15];
  const float* Wh1 = (const float*)d_in[16];
  const float* bh1 = (const float*)d_in[17];
  const float* gh  = (const float*)d_in[18];
  const float* bh  = (const float*)d_in[19];
  const float* Wh2 = (const float*)d_in[20];
  const float* bh2 = (const float*)d_in[21];
  float* out = (float*)d_out;

  char* ws = (char*)d_ws;
  float*     dinv   = (float*)(ws + O_DINV);
  _Float16*  P16    = (_Float16*)(ws + O_P16);
  _Float16*  X16    = (_Float16*)(ws + O_X16);
  _Float16*  Wa2h   = (_Float16*)(ws + O_WA2H);
  _Float16*  Ws2h   = (_Float16*)(ws + O_WS2H);
  float*     bias2  = (float*)(ws + O_B2);
  float*     nagg1c = (float*)(ws + O_NAGG1);
  _Float16*  h1c    = (_Float16*)(ws + O_H1);
  _Float16*  nagg2c = (_Float16*)(ws + O_NAGG2);
  float*     hmean  = (float*)(ws + O_HMEAN);

  k_prep_dinv<<<400, 64, 0, stream>>>(adj, dinv);
  k_prep_P<<<400, 64, 0, stream>>>(adj, dinv, P16);
  k_prep_x<<<4992, 256, 0, stream>>>(bold, X16);
  k_prep_w<<<1, 256, 0, stream>>>(Wa2, Ws2, ba2, bs2, Wa2h, Ws2h, bias2);

  for (int c = 0; c < 4; ++c) {
    k_l1gemm<<<30, 256, 0, stream>>>(X16, P16, nagg1c, c);
    k_l1pw<<<1200, 256, 0, stream>>>(bold, nagg1c, Wa1, ba1, Ws1, bs1, g1, b1,
                                     h1c, c);
    k_l2gemm<<<3840, 128, 0, stream>>>(h1c, P16, nagg2c);
    k_l2epi<<<400, 32, 0, stream>>>(nagg2c, h1c, Wa2h, Ws2h, bias2, g2, b2,
                                    hmean, c);
  }
  k_head<<<64, 256, 0, stream>>>(hmean, Wat, bat, Wh1, bh1, gh, bh, Wh2, bh2,
                                 out);
}
